// LocalImplicitSRNetRGBout_15985868275968
// MI455X (gfx1250) — matrix operand order repaired and checked by reading, not compiled, not run on hardware
//
#include <hip/hip_runtime.h>

// ---------------- problem constants ----------------
#define B_    4
#define C_    64
#define H_    48
#define W_    48
#define S_    2
#define Q_    ((H_*S_)*(W_*S_))   // 9216
#define BQ_   (B_*Q_)             // 36864
#define C9_   (C_*9)              // 576
#define KPAD_ 608                 // 580 padded to multiple of 32
#define HID_  256
#define EPS_  1e-6f

typedef __attribute__((ext_vector_type(16))) __bf16 v16bf;
typedef __attribute__((ext_vector_type(8)))  __bf16 v8bf;
typedef __attribute__((ext_vector_type(8)))  float  v8f;

// ---------------- device helpers ----------------
__device__ __forceinline__ int iclampi(int v, int lo, int hi) {
  return v < lo ? lo : (v > hi ? hi : v);
}

__device__ __forceinline__ void nearest_ij(float c0, float c1, int& iy, int& ix) {
  // jnp.round == rintf (round half to even)
  iy = iclampi((int)rintf(((c0 + 1.f) * (float)H_ - 1.f) * 0.5f), 0, H_ - 1);
  ix = iclampi((int)rintf(((c1 + 1.f) * (float)W_ - 1.f) * 0.5f), 0, W_ - 1);
}

__device__ __forceinline__ void offset_coord(float c0, float c1, float cl0, float cl1,
                                             int vx, int vy, float& o0, float& o1) {
  float rx = (1.f - cl0) / (float)(H_ - 1);   // 1/tx
  float ry = (1.f - cl1) / (float)(W_ - 1);   // 1/ty
  o0 = fminf(fmaxf(c0 + (float)vx * rx + EPS_, -1.f + 1e-6f), 1.f - 1e-6f);
  o1 = fminf(fmaxf(c1 + (float)vy * ry + EPS_, -1.f + 1e-6f), 1.f - 1e-6f);
}

// ---------------- weight conversion: fp32 [K x N] -> bf16 swizzled WMMA-B fragments ----
// fragment layout: idx = ((nt16*(K/32)+kb)*2 + hi)*256 + lo*16 + i, k = kb*32+hi*16+i, n = nt16*16+lo
__global__ void wswz_kernel(const float* __restrict__ src, __bf16* __restrict__ dst,
                            int Ksrc, int K, int N) {
  int t = blockIdx.x * blockDim.x + threadIdx.x;
  int total = K * N;
  if (t >= total) return;
  int i  = t & 15;
  int lo = (t >> 4) & 15;
  int hi = (t >> 8) & 1;
  int kbc = K >> 5;
  int kb = (t >> 9) % kbc;
  int nt = t / (kbc << 9);
  int k = kb * 32 + hi * 16 + i;
  int n = nt * 16 + lo;
  float v = (k < Ksrc) ? src[(size_t)k * N + n] : 0.f;
  dst[t] = (__bf16)v;
}

// ---------------- conv3x3 + bias ----------------
__global__ void conv3x3_kernel(const float* __restrict__ x, const float* __restrict__ w,
                               const float* __restrict__ bias, float* __restrict__ feat) {
  int t = blockIdx.x * blockDim.x + threadIdx.x;
  if (t >= B_ * C_ * H_ * W_) return;
  int xx = t % W_;
  int yy = (t / W_) % H_;
  int o  = (t / (H_ * W_)) % C_;
  int b  = t / (C_ * H_ * W_);
  float acc = bias[o];
  for (int i = 0; i < 3; ++i)
    for (int ky = 0; ky < 3; ++ky)
      for (int kx = 0; kx < 3; ++kx) {
        int y = yy + ky - 1, xq = xx + kx - 1;
        if (y >= 0 && y < H_ && xq >= 0 && xq < W_)
          acc += w[((o * 3 + i) * 3 + ky) * 3 + kx] *
                 x[((size_t)(b * 3 + i) * H_ + y) * W_ + xq];
      }
  feat[t] = acc;
}

// ---------------- gather: build MLP input rows (bf16, K padded to 608) ----------------
__global__ void gather_kernel(const float* __restrict__ feature, const float* __restrict__ coord,
                              const float* __restrict__ cell, __bf16* __restrict__ inp,
                              int vx, int vy) {
  int t = blockIdx.x * blockDim.x + threadIdx.x;
  if (t >= BQ_ * C_) return;
  int row = t / C_;
  int ch  = t - row * C_;
  int b   = row / Q_;
  float c0 = coord[(size_t)row * 2 + 0], c1 = coord[(size_t)row * 2 + 1];
  float cl0 = cell[(size_t)b * Q_ * 2 + 0], cl1 = cell[(size_t)b * Q_ * 2 + 1];
  float o0, o1;
  offset_coord(c0, c1, cl0, cl1, vx, vy, o0, o1);
  int iy, ix;
  nearest_ij(o0, o1, iy, ix);
  const float* f = feature + ((size_t)b * C_ + ch) * H_ * W_;
  __bf16* out = inp + (size_t)row * KPAD_ + ch * 9;
  for (int ki = 0; ki < 3; ++ki)
    for (int kj = 0; kj < 3; ++kj) {
      int y = iy + ki - 1, xq = ix + kj - 1;
      float v = (y >= 0 && y < H_ && xq >= 0 && xq < W_) ? f[y * W_ + xq] : 0.f;
      out[ki * 3 + kj] = (__bf16)v;
    }
  if (ch == 0) {  // write rel coords, scaled cell, and zero padding
    float ck0 = ((float)iy + 0.5f) * 2.f / (float)H_ - 1.f;
    float ck1 = ((float)ix + 0.5f) * 2.f / (float)W_ - 1.f;
    __bf16* tail = inp + (size_t)row * KPAD_ + C9_;
    tail[0] = (__bf16)((c0 - ck0) * (float)H_);
    tail[1] = (__bf16)((c1 - ck1) * (float)W_);
    tail[2] = (__bf16)(cl0 * (float)H_);
    tail[3] = (__bf16)(cl1 * (float)W_);
    for (int i = 4; i < KPAD_ - C9_; ++i) tail[i] = (__bf16)0.f;
  }
}

// ---------------- WMMA GEMM: C[MxN] = act(A[MxK] * Bswz + bias) ----------------
// Register-blocked: one wave computes a 32x64 tile = 2 (M) x 4 (N) WMMA sub-tiles.
// Per K-step: 2 A-fragment loads + 4 B-fragment loads feed 8 v_wmma ops
// (~0.75 KB of loads per WMMA -> ~21 FLOP/byte vs 8 for the naive 16x16 tile).
template <int RELU, int OUT_BF16>
__global__ __launch_bounds__(128)
void gemm_bf16_wmma(const __bf16* __restrict__ A, const __bf16* __restrict__ Bswz,
                    const float* __restrict__ bias, void* __restrict__ Cout,
                    int M, int N, int K) {
  const int wave   = blockIdx.x * 4 + (threadIdx.x >> 5);
  const int tilesN = N >> 6;                 // 64-wide column tiles
  const int mt = wave / tilesN;              // 32-row tile index
  const int nt = wave - mt * tilesN;
  if (mt * 32 >= M) return;
  const int lane = threadIdx.x & 31;
  const int hi = lane >> 4;
  const int lo = lane & 15;
  const int kbc = K >> 5;
  const __bf16* Arow0 = A + (size_t)(mt * 32 + lo) * K;
  const __bf16* Arow1 = Arow0 + (size_t)16 * K;
  // B fragment pointer for N-subtile 0; subtiles are kbc*512 elements apart
  const __bf16* Bp0 = Bswz + ((((size_t)(nt * 4) * kbc) * 2 + hi) * 16 + lo) * 16;
  const size_t bsub = (size_t)kbc * 512;

  v8f acc[2][4] = {};
  for (int kb = 0; kb < kbc; ++kb) {
    v8bf a00 = *(const v8bf*)(Arow0 + kb * 32 + hi * 16);
    v8bf a01 = *(const v8bf*)(Arow0 + kb * 32 + hi * 16 + 8);
    v8bf a10 = *(const v8bf*)(Arow1 + kb * 32 + hi * 16);
    v8bf a11 = *(const v8bf*)(Arow1 + kb * 32 + hi * 16 + 8);
    v16bf a0, a1;
#pragma unroll
    for (int i = 0; i < 8; ++i) {
      a0[i] = a00[i]; a0[8 + i] = a01[i];
      a1[i] = a10[i]; a1[8 + i] = a11[i];
    }
#pragma unroll
    for (int n = 0; n < 4; ++n) {
      v16bf bfrag = *(const v16bf*)(Bp0 + n * bsub + (size_t)kb * 512);
      acc[0][n] = __builtin_amdgcn_wmma_f32_16x16x32_bf16(false, a0, false, bfrag,
                                                          (short)0, acc[0][n], false, false);
      acc[1][n] = __builtin_amdgcn_wmma_f32_16x16x32_bf16(false, a1, false, bfrag,
                                                          (short)0, acc[1][n], false, false);
    }
  }
#pragma unroll
  for (int m = 0; m < 2; ++m) {
#pragma unroll
    for (int n = 0; n < 4; ++n) {
      const int col = (nt * 4 + n) * 16 + lo;
      const float bv = bias[col];
#pragma unroll
      for (int i = 0; i < 8; ++i) {
        const int row = mt * 32 + m * 16 + hi * 8 + i;
        float v = acc[m][n][i] + bv;
        if (RELU) v = fmaxf(v, 0.f);
        if (OUT_BF16) ((__bf16*)Cout)[(size_t)row * N + col] = (__bf16)v;
        else          ((float*)Cout)[(size_t)row * N + col] = v;
      }
    }
  }
}

// ---------------- attn_j = sum_c query_c * kv_c * wk_c  (wave per row) ----------------
__global__ __launch_bounds__(256)
void attn_kernel(const float* __restrict__ feature, const float* __restrict__ coord,
                 const float* __restrict__ cell, const float* __restrict__ wk,
                 float* __restrict__ attn, int vx, int vy, int j) {
  int row = blockIdx.x * 8 + (threadIdx.x >> 5);
  if (row >= BQ_) return;
  int lane = threadIdx.x & 31;
  int b = row / Q_;
  float c0 = coord[(size_t)row * 2 + 0], c1 = coord[(size_t)row * 2 + 1];
  float cl0 = cell[(size_t)b * Q_ * 2 + 0], cl1 = cell[(size_t)b * Q_ * 2 + 1];
  int iyq, ixq;
  nearest_ij(c0, c1, iyq, ixq);
  float o0, o1;
  offset_coord(c0, c1, cl0, cl1, vx, vy, o0, o1);
  int iy, ix;
  nearest_ij(o0, o1, iy, ix);
  const float* fb = feature + (size_t)b * C_ * H_ * W_;
  float sum = 0.f;
  for (int c = lane; c < C9_; c += 32) {
    int ch = c / 9, kk = c - ch * 9;
    int ki = kk / 3, kj = kk - ki * 3;
    const float* f = fb + (size_t)ch * H_ * W_;
    int yq = iyq + ki - 1, xq = ixq + kj - 1;
    int yk = iy + ki - 1, xk = ix + kj - 1;
    float qv = (yq >= 0 && yq < H_ && xq >= 0 && xq < W_) ? f[yq * W_ + xq] : 0.f;
    float kv = (yk >= 0 && yk < H_ && xk >= 0 && xk < W_) ? f[yk * W_ + xk] : 0.f;
    sum += qv * kv * wk[(size_t)row * C9_ + c];
  }
#pragma unroll
  for (int m = 16; m >= 1; m >>= 1) sum += __shfl_xor(sum, m, 32);
  if (lane == 0) attn[(size_t)row * 4 + j] = sum;
}

// ---------------- softmax over 4 offsets, in place ----------------
__global__ void softmax4_kernel(float* __restrict__ attn) {
  int row = blockIdx.x * blockDim.x + threadIdx.x;
  if (row >= BQ_) return;
  float a0 = attn[row * 4 + 0], a1 = attn[row * 4 + 1];
  float a2 = attn[row * 4 + 2], a3 = attn[row * 4 + 3];
  float m = fmaxf(fmaxf(a0, a1), fmaxf(a2, a3));
  float e0 = __expf(a0 - m), e1 = __expf(a1 - m), e2 = __expf(a2 - m), e3 = __expf(a3 - m);
  float inv = 1.f / (e0 + e1 + e2 + e3);
  attn[row * 4 + 0] = e0 * inv; attn[row * 4 + 1] = e1 * inv;
  attn[row * 4 + 2] = e2 * inv; attn[row * 4 + 3] = e3 * inv;
}

// ---------------- xf (+)= a_j * kv_c * wv_c  (wave per row) ----------------
__global__ __launch_bounds__(256)
void accum_kernel(const float* __restrict__ feature, const float* __restrict__ coord,
                  const float* __restrict__ cell, const float* __restrict__ wv,
                  const float* __restrict__ attn, float* __restrict__ xf,
                  int vx, int vy, int j, int first) {
  int row = blockIdx.x * 8 + (threadIdx.x >> 5);
  if (row >= BQ_) return;
  int lane = threadIdx.x & 31;
  int b = row / Q_;
  float c0 = coord[(size_t)row * 2 + 0], c1 = coord[(size_t)row * 2 + 1];
  float cl0 = cell[(size_t)b * Q_ * 2 + 0], cl1 = cell[(size_t)b * Q_ * 2 + 1];
  float o0, o1;
  offset_coord(c0, c1, cl0, cl1, vx, vy, o0, o1);
  int iy, ix;
  nearest_ij(o0, o1, iy, ix);
  float a = attn[(size_t)row * 4 + j];
  const float* fb = feature + (size_t)b * C_ * H_ * W_;
  for (int c = lane; c < C9_; c += 32) {
    int ch = c / 9, kk = c - ch * 9;
    int ki = kk / 3, kj = kk - ki * 3;
    const float* f = fb + (size_t)ch * H_ * W_;
    int yk = iy + ki - 1, xk = ix + kj - 1;
    float kv = (yk >= 0 && yk < H_ && xk >= 0 && xk < W_) ? f[yk * W_ + xk] : 0.f;
    float val = a * kv * wv[(size_t)row * C9_ + c];
    size_t idx = (size_t)row * C9_ + c;
    xf[idx] = first ? val : (xf[idx] + val);
  }
}

// ---------------- fp32 -> bf16 ----------------
__global__ void f32_to_bf16_kernel(const float* __restrict__ src, __bf16* __restrict__ dst, int n) {
  int t = blockIdx.x * blockDim.x + threadIdx.x;
  if (t < n) dst[t] = (__bf16)src[t];
}

// ---------------- head: out = act[row,:] . w + b  (wave per row) ----------------
__global__ __launch_bounds__(256)
void head_kernel(const __bf16* __restrict__ act, const float* __restrict__ w,
                 const float* __restrict__ b, float* __restrict__ head_out, int hidx) {
  int row = blockIdx.x * 8 + (threadIdx.x >> 5);
  if (row >= BQ_) return;
  int lane = threadIdx.x & 31;
  float sum = 0.f;
#pragma unroll
  for (int c = lane; c < HID_; c += 32)
    sum += (float)act[(size_t)row * HID_ + c] * w[c];
#pragma unroll
  for (int m = 16; m >= 1; m >>= 1) sum += __shfl_xor(sum, m, 32);
  if (lane == 0) head_out[(size_t)row * 3 + hidx] = sum + b[0];
}

// ---------------- final: pred + bilinear-border sample of x ----------------
__global__ void final_kernel(const float* __restrict__ x, const float* __restrict__ coord,
                             const float* __restrict__ head_out, float* __restrict__ out) {
  int row = blockIdx.x * blockDim.x + threadIdx.x;
  if (row >= BQ_) return;
  int b = row / Q_;
  float c0 = coord[(size_t)row * 2 + 0], c1 = coord[(size_t)row * 2 + 1];
  float fy = ((c0 + 1.f) * (float)H_ - 1.f) * 0.5f;
  float fx = ((c1 + 1.f) * (float)W_ - 1.f) * 0.5f;
  float y0f = floorf(fy), x0f = floorf(fx);
  float wy = fy - y0f, wx = fx - x0f;
  int y0 = iclampi((int)y0f, 0, H_ - 1), y1 = iclampi((int)y0f + 1, 0, H_ - 1);
  int x0 = iclampi((int)x0f, 0, W_ - 1), x1 = iclampi((int)x0f + 1, 0, W_ - 1);
  for (int ch = 0; ch < 3; ++ch) {
    const float* img = x + ((size_t)(b * 3 + ch) * H_) * W_;
    float v = img[y0 * W_ + x0] * (1.f - wy) * (1.f - wx)
            + img[y0 * W_ + x1] * (1.f - wy) * wx
            + img[y1 * W_ + x0] * wy * (1.f - wx)
            + img[y1 * W_ + x1] * wy * wx;
    out[(size_t)row * 3 + ch] = head_out[(size_t)row * 3 + ch] + v;
  }
}

// =======================================================================
extern "C" void kernel_launch(void* const* d_in, const int* in_sizes, int n_in,
                              void* d_out, int out_size, void* d_ws, size_t ws_size,
                              hipStream_t stream) {
  (void)in_sizes; (void)n_in; (void)out_size; (void)ws_size;
  // ---- inputs (setup_inputs order, pytree-flattened) ----
  const float* x      = (const float*)d_in[0];
  const float* coord  = (const float*)d_in[1];
  const float* cell   = (const float*)d_in[2];
  const float* conv_w = (const float*)d_in[3];
  const float* conv_b = (const float*)d_in[4];
  // params_qr: 5..14, params_qg: 15..24, params_qb: 25..34, params_k: 35..44, params_v: 45..54
  const int QRB = 5, QGB = 15, QBB = 25, KB = 35, VB = 45;
  auto Wf = [&](int base, int l) { return (const float*)d_in[base + 2 * l]; };
  auto Bf = [&](int base, int l) { return (const float*)d_in[base + 2 * l + 1]; };

  // ---- workspace carve-up ----
  size_t off = 0;
  auto carve = [&](size_t bytes) -> char* {
    char* p = (char*)d_ws + off;
    off = (off + bytes + 255) & ~(size_t)255;
    return p;
  };
  float*  feat   = (float*) carve((size_t)B_ * C_ * H_ * W_ * 4);
  __bf16* inp    = (__bf16*)carve((size_t)BQ_ * KPAD_ * 2);       // also reused as xf_bf16
  __bf16* actA   = (__bf16*)carve((size_t)BQ_ * HID_ * 2);
  __bf16* actB   = (__bf16*)carve((size_t)BQ_ * HID_ * 2);
  float*  wbuf   = (float*) carve((size_t)BQ_ * C9_ * 4);
  float*  xf     = (float*) carve((size_t)BQ_ * C9_ * 4);
  float*  attn   = (float*) carve((size_t)BQ_ * 4 * 4);
  float*  hout   = (float*) carve((size_t)BQ_ * 3 * 4);
  __bf16* WK[5], *WV[5], *WQ[3][4];
  const size_t szL0kv = (size_t)KPAD_ * HID_ * 2;
  const size_t szHid  = (size_t)HID_ * HID_ * 2;
  const size_t szL4   = (size_t)HID_ * C9_ * 2;
  const size_t szL0q  = (size_t)C9_ * HID_ * 2;
  WK[0] = (__bf16*)carve(szL0kv);
  for (int l = 1; l <= 3; ++l) WK[l] = (__bf16*)carve(szHid);
  WK[4] = (__bf16*)carve(szL4);
  WV[0] = (__bf16*)carve(szL0kv);
  for (int l = 1; l <= 3; ++l) WV[l] = (__bf16*)carve(szHid);
  WV[4] = (__bf16*)carve(szL4);
  for (int h = 0; h < 3; ++h) {
    WQ[h][0] = (__bf16*)carve(szL0q);
    for (int l = 1; l <= 3; ++l) WQ[h][l] = (__bf16*)carve(szHid);
  }

  auto cvt = [&](const float* src, __bf16* dst, int Ksrc, int K, int N) {
    int total = K * N;
    wswz_kernel<<<(total + 255) / 256, 256, 0, stream>>>(src, dst, Ksrc, K, N);
  };

  // ---- 1. weights -> bf16 swizzled fragments ----
  cvt(Wf(KB, 0), WK[0], C9_ + 4, KPAD_, HID_);
  cvt(Wf(VB, 0), WV[0], C9_ + 4, KPAD_, HID_);
  for (int l = 1; l <= 3; ++l) {
    cvt(Wf(KB, l), WK[l], HID_, HID_, HID_);
    cvt(Wf(VB, l), WV[l], HID_, HID_, HID_);
  }
  cvt(Wf(KB, 4), WK[4], HID_, HID_, C9_);
  cvt(Wf(VB, 4), WV[4], HID_, HID_, C9_);
  const int qbases[3] = {QRB, QGB, QBB};
  for (int h = 0; h < 3; ++h) {
    cvt(Wf(qbases[h], 0), WQ[h][0], C9_, C9_, HID_);
    for (int l = 1; l <= 3; ++l) cvt(Wf(qbases[h], l), WQ[h][l], HID_, HID_, HID_);
  }

  // ---- 2. conv3x3 ----
  conv3x3_kernel<<<(B_ * C_ * H_ * W_ + 255) / 256, 256, 0, stream>>>(x, conv_w, conv_b, feat);

  const int M = BQ_;
  // one wave per 32x64 tile, 4 waves per block
  const int gB256 = (M / 32) * (HID_ / 64) / 4;   // N=256 GEMMs
  const int gB576 = (M / 32) * (C9_ / 64) / 4;    // N=576 GEMMs
  auto run_mlp = [&](__bf16** Wl, int base, float* outBuf) {
    gemm_bf16_wmma<1, 1><<<gB256, 128, 0, stream>>>(inp,  Wl[0], Bf(base, 0), actA, M, HID_, KPAD_);
    gemm_bf16_wmma<1, 1><<<gB256, 128, 0, stream>>>(actA, Wl[1], Bf(base, 1), actB, M, HID_, HID_);
    gemm_bf16_wmma<1, 1><<<gB256, 128, 0, stream>>>(actB, Wl[2], Bf(base, 2), actA, M, HID_, HID_);
    gemm_bf16_wmma<1, 1><<<gB256, 128, 0, stream>>>(actA, Wl[3], Bf(base, 3), actB, M, HID_, HID_);
    gemm_bf16_wmma<0, 0><<<gB576, 128, 0, stream>>>(actB, Wl[4], Bf(base, 4), outBuf, M, C9_, HID_);
  };

  const int vlist[4][2] = {{-1, -1}, {-1, 1}, {1, -1}, {1, 1}};
  const int gGather = (BQ_ * C_ + 255) / 256;
  const int gWaveRow = (BQ_ + 7) / 8;

  // ---- 3. pass K: attention logits ----
  for (int j = 0; j < 4; ++j) {
    gather_kernel<<<gGather, 256, 0, stream>>>(feat, coord, cell, inp, vlist[j][0], vlist[j][1]);
    run_mlp(WK, KB, wbuf);
    attn_kernel<<<gWaveRow, 256, 0, stream>>>(feat, coord, cell, wbuf, attn,
                                              vlist[j][0], vlist[j][1], j);
  }

  // ---- 4. softmax ----
  softmax4_kernel<<<(BQ_ + 255) / 256, 256, 0, stream>>>(attn);

  // ---- 5. pass V: weighted value accumulation ----
  for (int j = 0; j < 4; ++j) {
    gather_kernel<<<gGather, 256, 0, stream>>>(feat, coord, cell, inp, vlist[j][0], vlist[j][1]);
    run_mlp(WV, VB, wbuf);
    accum_kernel<<<gWaveRow, 256, 0, stream>>>(feat, coord, cell, wbuf, attn, xf,
                                               vlist[j][0], vlist[j][1], j, j == 0 ? 1 : 0);
  }

  // ---- 6. xf -> bf16 (reuse inp buffer, row stride C9_) ----
  __bf16* xf16 = inp;
  f32_to_bf16_kernel<<<((BQ_ * C9_) + 255) / 256, 256, 0, stream>>>(xf, xf16, BQ_ * C9_);

  // ---- 7. qr/qg/qb chains ----
  for (int h = 0; h < 3; ++h) {
    int base = qbases[h];
    gemm_bf16_wmma<1, 1><<<gB256, 128, 0, stream>>>(xf16, WQ[h][0], Bf(base, 0), actA, M, HID_, C9_);
    gemm_bf16_wmma<1, 1><<<gB256, 128, 0, stream>>>(actA, WQ[h][1], Bf(base, 1), actB, M, HID_, HID_);
    gemm_bf16_wmma<1, 1><<<gB256, 128, 0, stream>>>(actB, WQ[h][2], Bf(base, 2), actA, M, HID_, HID_);
    gemm_bf16_wmma<1, 1><<<gB256, 128, 0, stream>>>(actA, WQ[h][3], Bf(base, 3), actB, M, HID_, HID_);
    head_kernel<<<gWaveRow, 256, 0, stream>>>(actB, Wf(base, 4), Bf(base, 4), hout, h);
  }

  // ---- 8. residual + output ----
  final_kernel<<<(BQ_ + 255) / 256, 256, 0, stream>>>(x, coord, hout, (float*)d_out);
}
